// DeepseekBlock_80547816669400
// MI455X (gfx1250) — matrix operand order repaired and checked by reading, not compiled, not run on hardware
//
#include <hip/hip_runtime.h>

// ---------------------------------------------------------------------------
// Types / helpers for CDNA5 WMMA (wave32, 16x16x32 bf16 -> f32)
// ---------------------------------------------------------------------------
typedef __attribute__((ext_vector_type(16))) __bf16   v16bf;
typedef __attribute__((ext_vector_type(8)))  __bf16   v8bf;
typedef __attribute__((ext_vector_type(8)))  float    v8f;
typedef __attribute__((ext_vector_type(4)))  unsigned v4u;
typedef unsigned short u16;

#define DEV static __device__ __forceinline__

DEV u16 f2bfu(float f) {                         // f32 -> bf16 bits, RNE
  unsigned u = __builtin_bit_cast(unsigned, f);
  u += 0x7FFFu + ((u >> 16) & 1u);
  return (u16)(u >> 16);
}
DEV v8bf ld8(const u16* p) {                     // 8 consecutive bf16 (16B)
  return __builtin_bit_cast(v8bf, *(const v4u*)p);
}
DEV v16bf cat16(v8bf lo, v8bf hi) {
  return __builtin_shufflevector(lo, hi, 0,1,2,3,4,5,6,7,8,9,10,11,12,13,14,15);
}
DEV v8f wmma_bf16(v16bf a, v16bf b, v8f c) {
  return __builtin_amdgcn_wmma_f32_16x16x32_bf16(false, a, false, b, (short)0, c,
                                                 false, false);
}

// LDS byte offset of a __shared__ object (generic -> AS(3) -> int)
DEV unsigned lds_addr(const void* p) {
  return (unsigned)(unsigned long long)(__attribute__((address_space(3))) const void*)p;
}
// CDNA5 async memory->LDS copy, 16B per lane (ASYNCcnt path)
DEV void async_b128(unsigned lds_off, const u16* gaddr) {
  asm volatile("global_load_async_to_lds_b128 %0, %1, off"
               :: "v"(lds_off), "v"(gaddr) : "memory");
}
#define WAIT_ASYNC(n) asm volatile("s_wait_asynccnt " #n ::: "memory")

// DPP row-reduction within 16-lane halves (no LDS/bpermute traffic)
template <int CTRL>
DEV float dppmov(float x) {
  return __builtin_bit_cast(
      float, __builtin_amdgcn_mov_dpp(__builtin_bit_cast(int, x), CTRL, 0xf, 0xf,
                                      true));
}
DEV float redmax16(float v) {
  v = fmaxf(v, dppmov<0xB1>(v));   // quad_perm xor1
  v = fmaxf(v, dppmov<0x4E>(v));   // quad_perm xor2
  v = fmaxf(v, dppmov<0x141>(v));  // row_half_mirror (4 -> 8)
  v = fmaxf(v, dppmov<0x140>(v));  // row_mirror      (8 -> 16)
  return v;
}
DEV float redsum16(float v) {
  v += dppmov<0xB1>(v);
  v += dppmov<0x4E>(v);
  v += dppmov<0x141>(v);
  v += dppmov<0x140>(v);
  return v;
}

// ---------------------------------------------------------------------------
// Weight convert+transpose: W (K x N, f32) -> Wt (N x K, bf16). 32x32 tiles.
// ---------------------------------------------------------------------------
__global__ __launch_bounds__(256) void transcvt_k(const float* __restrict__ W,
                                                  u16* __restrict__ Wt,
                                                  int K, int N) {
  __shared__ float tile[32][33];
  W  += (size_t)blockIdx.z * K * N;
  Wt += (size_t)blockIdx.z * K * N;
  const int kb = blockIdx.y * 32, nb = blockIdx.x * 32;
  const int tx = threadIdx.x & 31, ty = threadIdx.x >> 5;   // 32 x 8
#pragma unroll
  for (int i = 0; i < 4; ++i)
    tile[ty + i * 8][tx] = W[(size_t)(kb + ty + i * 8) * N + nb + tx];
  __syncthreads();
#pragma unroll
  for (int i = 0; i < 4; ++i)
    Wt[(size_t)(nb + ty + i * 8) * K + kb + tx] = f2bfu(tile[tx][ty + i * 8]);
}

// ---------------------------------------------------------------------------
// RMSNorm row of 1024: optional f32 and/or bf16 outputs.
// ---------------------------------------------------------------------------
__global__ __launch_bounds__(256) void rmsnorm_k(const float* __restrict__ X,
                                                 const float* __restrict__ W,
                                                 float* __restrict__ Yf,
                                                 u16* __restrict__ Yb) {
  __shared__ float red[8];
  __shared__ float rs_s;
  const int row = blockIdx.x;
  const float* x = X + (size_t)row * 1024;
  float s = 0.f;
#pragma unroll
  for (int i = 0; i < 4; ++i) { float v = x[threadIdx.x + i * 256]; s += v * v; }
#pragma unroll
  for (int off = 16; off; off >>= 1) s += __shfl_xor(s, off, 32);
  if ((threadIdx.x & 31) == 0) red[threadIdx.x >> 5] = s;
  __syncthreads();
  if (threadIdx.x == 0) {
    float t = 0.f;
#pragma unroll
    for (int i = 0; i < 8; ++i) t += red[i];
    rs_s = rsqrtf(t / 1024.f + 1e-6f);
  }
  __syncthreads();
  const float rs = rs_s;
#pragma unroll
  for (int i = 0; i < 4; ++i) {
    int c = threadIdx.x + i * 256;
    float y = x[c] * rs * W[c];
    if (Yf) Yf[(size_t)row * 1024 + c] = y;
    if (Yb) Yb[(size_t)row * 1024 + c] = f2bfu(y);
  }
}

// ---------------------------------------------------------------------------
// bf16-WMMA GEMM: C(MxN) = A(MxK,bf16) @ Bt(NxK,bf16 pre-transposed).
//   EPI 0: C=acc  EPI 1: C=acc+R  EPI 2: C += gate[row]*acc
//   OBF 0: f32 C only  OBF 1: bf16 Cb only  OBF 2: both
// Block 256 thr (8 waves); tile 64x64x32; per-wave 16x32 (2 WMMA / k-step).
// Double-buffered LDS, async global->LDS staging (ASYNCcnt), 16B per thread.
// ---------------------------------------------------------------------------
template <int EPI, int OBF>
__global__ __launch_bounds__(256) void gemm_k(
    const u16* __restrict__ A, const u16* __restrict__ Bt,
    float* __restrict__ C, u16* __restrict__ Cb, const float* __restrict__ R,
    const float* __restrict__ G, int gstride, int M, int N, int Kdim) {
  __shared__ u16 As[2][64 * 32];   // [m][k]
  __shared__ u16 Bs[2][64 * 32];   // [n][k]
  const int tid  = threadIdx.x;
  const int lane = tid & 31;
  const int wave = tid >> 5;
  const int rih  = lane & 15;
  const int half = lane >> 4;
  const int m0 = (wave & 3) * 16;
  const int n0 = (wave >> 2) * 32;
  const int blockM = blockIdx.y * 64;
  const int blockN = blockIdx.x * 64;

  // staging: each thread owns one 16B chunk of the A tile and the B tile
  const int sr = tid >> 2, sc = (tid & 3) * 8;
  const u16* ag = A  + (size_t)(blockM + sr) * Kdim + sc;
  const u16* bg = Bt + (size_t)(blockN + sr) * Kdim + sc;
  const unsigned aoff[2] = { lds_addr(&As[0][sr * 32 + sc]),
                             lds_addr(&As[1][sr * 32 + sc]) };
  const unsigned boff[2] = { lds_addr(&Bs[0][sr * 32 + sc]),
                             lds_addr(&Bs[1][sr * 32 + sc]) };

  v8f acc0 = {}; v8f acc1 = {};
  const int nk = Kdim >> 5;
  async_b128(aoff[0], ag);          // prologue: tile 0 -> buffer 0
  async_b128(boff[0], bg);
  for (int t = 0; t < nk; ++t) {
    const int cur = t & 1;
    if (t + 1 < nk) {               // overlap next tile with this compute
      async_b128(aoff[cur ^ 1], ag + (size_t)(t + 1) * 32);
      async_b128(boff[cur ^ 1], bg + (size_t)(t + 1) * 32);
      WAIT_ASYNC(0x2);              // tile t complete (in-order), t+1 in flight
    } else {
      WAIT_ASYNC(0x0);
    }
    __syncthreads();
    const u16* ap  = &As[cur][(m0 + rih) * 32 + half * 8];
    const u16* bp0 = &Bs[cur][(n0 + rih) * 32 + half * 8];
    const u16* bp1 = &Bs[cur][(n0 + 16 + rih) * 32 + half * 8];
    v16bf af  = cat16(ld8(ap), ld8(ap + 16));      // A: {kb..kb+7}U{kb+16..}
    v16bf bf0 = cat16(ld8(bp0), ld8(bp0 + 16));    // B: {kb..kb+7}U{kb+16..}
    v16bf bf1 = cat16(ld8(bp1), ld8(bp1 + 16));
    acc0 = wmma_bf16(af, bf0, acc0);
    acc1 = wmma_bf16(af, bf1, acc1);
    __syncthreads();
  }
#pragma unroll
  for (int r = 0; r < 8; ++r) {
    int row = blockM + m0 + half * 8 + r;
    size_t base = (size_t)row * N;
    int c0 = blockN + n0 + rih;
    float v0 = acc0[r], v1 = acc1[r];
    if constexpr (EPI == 1) { v0 += R[base + c0]; v1 += R[base + c0 + 16]; }
    if constexpr (EPI == 2) {
      float g = G[(size_t)row * gstride];
      v0 = C[base + c0] + g * v0;
      v1 = C[base + c0 + 16] + g * v1;
    }
    if constexpr (OBF != 1) { C[base + c0] = v0; C[base + c0 + 16] = v1; }
    if constexpr (OBF >= 1) {
      Cb[base + c0] = f2bfu(v0);
      Cb[base + c0 + 16] = f2bfu(v1);
    }
  }
}

// ---------------------------------------------------------------------------
// Flash attention (causal), all-bf16 operands. 4 waves/block share one head;
// K chunk staged via async-to-LDS, V chunk transposed via scatter; DPP row
// reductions for the online softmax. Output written bf16 for the Wo GEMM.
// ---------------------------------------------------------------------------
__global__ __launch_bounds__(128) void flash_k(const u16* __restrict__ Qb,
                                               const u16* __restrict__ Kb,
                                               const u16* __restrict__ Vb,
                                               u16* __restrict__ Oatt) {
  __shared__ u16 Ks[32 * 64];      // [key][dh]
  __shared__ u16 Vt[64 * 32];      // [dh][key]
  __shared__ u16 Pls[4 * 16 * 32]; // per-wave P tile
  const int tid = threadIdx.x, lane = tid & 31, wv = tid >> 5;
  const int rih = lane & 15, half = lane >> 4;
  const int h = blockIdx.y;
  const int q0blk = blockIdx.x * 64;
  const int q0 = q0blk + wv * 16;

  // Q fragments: direct aligned vector loads from bf16 global
  v16bf qf[2];
  {
    const u16* qp = Qb + (size_t)(q0 + rih) * 1024 + h * 64 + half * 8;
#pragma unroll
    for (int c = 0; c < 2; ++c)
      qf[c] = cat16(ld8(qp + c * 32), ld8(qp + c * 32 + 16));
  }

  float m_r[8], l_r[8], alph[8];
  v8f o[4] = {};
#pragma unroll
  for (int r = 0; r < 8; ++r) { m_r[r] = -1e30f; l_r[r] = 0.f; }

  u16* P = Pls + wv * (16 * 32);
  const u16* pp = P + rih * 32 + half * 8;
  const int sr = tid >> 2, sc = (tid & 3) * 16;     // staging: 32 x 64
  const unsigned ks0 = lds_addr(&Ks[sr * 64 + sc]);
  const unsigned ks1 = lds_addr(&Ks[sr * 64 + sc + 8]);
  const u16* kgb = Kb + (size_t)sr * 1024 + h * 64 + sc;
  const u16* vgb = Vb + (size_t)sr * 1024 + h * 64 + sc;
  const int kend = q0blk + 63;

  for (int kb = 0; kb <= kend; kb += 32) {
    // ---- async K chunk -> LDS; V chunk loaded + scattered transposed -----
    const u16* kp = kgb + (size_t)kb * 1024;
    async_b128(ks0, kp);
    async_b128(ks1, kp + 8);
    const u16* vp = vgb + (size_t)kb * 1024;
#pragma unroll
    for (int i = 0; i < 16; ++i) Vt[(sc + i) * 32 + sr] = vp[i];
    WAIT_ASYNC(0x0);
    __syncthreads();
    // ---- scores: 16 q x 32 k ---------------------------------------------
    v8f s0 = {}, s1 = {};
#pragma unroll
    for (int g = 0; g < 2; ++g) {
      const u16* kr = &Ks[(g * 16 + rih) * 64 + half * 8];
#pragma unroll
      for (int c = 0; c < 2; ++c) {
        v16bf bf = cat16(ld8(kr + c * 32), ld8(kr + c * 32 + 16));
        if (g == 0) s0 = wmma_bf16(qf[c], bf, s0);
        else        s1 = wmma_bf16(qf[c], bf, s1);
      }
    }
    // ---- causal mask + online softmax (DPP reductions) -------------------
#pragma unroll
    for (int r = 0; r < 8; ++r) {
      const int row = q0 + half * 8 + r;
      float a = (kb + rih <= row)      ? s0[r] * 0.125f : -1e30f;
      float b = (kb + 16 + rih <= row) ? s1[r] * 0.125f : -1e30f;
      const float mnew = fmaxf(m_r[r], redmax16(fmaxf(a, b)));
      alph[r] = __expf(m_r[r] - mnew);
      const float e0 = __expf(a - mnew), e1 = __expf(b - mnew);
      l_r[r] = l_r[r] * alph[r] + redsum16(e0 + e1);
      m_r[r] = mnew;
      const int prow = half * 8 + r;
      P[prow * 32 + rih]      = f2bfu(e0);
      P[prow * 32 + 16 + rih] = f2bfu(e1);
    }
#pragma unroll
    for (int b = 0; b < 4; ++b)
#pragma unroll
      for (int r = 0; r < 8; ++r) o[b][r] *= alph[r];
    // ---- O += P @ V ------------------------------------------------------
    v16bf pf = cat16(ld8(pp), ld8(pp + 16));
#pragma unroll
    for (int b = 0; b < 4; ++b) {
      const u16* vr = &Vt[(b * 16 + rih) * 32 + half * 8];
      v16bf vf = cat16(ld8(vr), ld8(vr + 16));
      o[b] = wmma_bf16(pf, vf, o[b]);
    }
    __syncthreads();
  }
#pragma unroll
  for (int b = 0; b < 4; ++b)
#pragma unroll
    for (int r = 0; r < 8; ++r) {
      const int row = q0 + half * 8 + r;
      Oatt[(size_t)row * 1024 + h * 64 + b * 16 + rih] = f2bfu(o[b][r] / l_r[r]);
    }
}

// ---------------------------------------------------------------------------
// Router: softmax over E=8, top-2 renormalized gates, aux stats. Wave/token.
// ---------------------------------------------------------------------------
__global__ __launch_bounds__(256) void router_k(const float* __restrict__ xn,
                                                const float* __restrict__ Wr,
                                                float* __restrict__ gatef,
                                                float* __restrict__ cnt,
                                                float* __restrict__ psum) {
  const int lane = threadIdx.x & 31;
  const int t = blockIdx.x * 8 + (threadIdx.x >> 5);
  const float* x = xn + (size_t)t * 1024;
  float acc[8] = {};
  for (int d = lane; d < 1024; d += 32) {
    const float xv = x[d];
    const float* wr = Wr + (size_t)d * 8;
#pragma unroll
    for (int e = 0; e < 8; ++e) acc[e] += xv * wr[e];
  }
#pragma unroll
  for (int e = 0; e < 8; ++e)
#pragma unroll
    for (int off = 16; off; off >>= 1) acc[e] += __shfl_xor(acc[e], off, 32);
  if (lane == 0) {
    float mx = acc[0];
#pragma unroll
    for (int e = 1; e < 8; ++e) mx = fmaxf(mx, acc[e]);
    float p[8], sum = 0.f;
#pragma unroll
    for (int e = 0; e < 8; ++e) { p[e] = __expf(acc[e] - mx); sum += p[e]; }
#pragma unroll
    for (int e = 0; e < 8; ++e) p[e] /= sum;
    int i0 = 0;
#pragma unroll
    for (int e = 1; e < 8; ++e) if (p[e] > p[i0]) i0 = e;
    int i1 = (i0 == 0) ? 1 : 0;
#pragma unroll
    for (int e = 0; e < 8; ++e) if (e != i0 && p[e] > p[i1]) i1 = e;
    const float gs = p[i0] + p[i1];
#pragma unroll
    for (int e = 0; e < 8; ++e) gatef[(size_t)t * 8 + e] = 0.f;
    gatef[(size_t)t * 8 + i0] = p[i0] / gs;
    gatef[(size_t)t * 8 + i1] = p[i1] / gs;
    atomicAdd(&cnt[i0], 1.f);
    atomicAdd(&cnt[i1], 1.f);
#pragma unroll
    for (int e = 0; e < 8; ++e) atomicAdd(&psum[e], p[e]);
  }
}

// ---------------------------------------------------------------------------
__global__ __launch_bounds__(256) void swiglu_k(const float* __restrict__ a,
                                                const float* __restrict__ b,
                                                u16* __restrict__ h, int n) {
  int i = blockIdx.x * 256 + threadIdx.x;
  if (i < n) { float x = a[i]; h[i] = f2bfu((x / (1.f + __expf(-x))) * b[i]); }
}
__global__ __launch_bounds__(256) void add_k(const float* __restrict__ a,
                                             const float* __restrict__ b,
                                             float* __restrict__ o, int n) {
  int i = blockIdx.x * 256 + threadIdx.x;
  if (i < n) o[i] = a[i] + b[i];
}
__global__ void aux_k(const float* cnt, const float* psum, float* out) {
  if (threadIdx.x == 0 && blockIdx.x == 0) {
    const float T = 2048.f;
    float aux = 0.f;
    for (int e = 0; e < 8; ++e)
      aux += (cnt[e] * (8.f / (2.f * T))) * (psum[e] / T);
    out[0] = aux * 8.f;
  }
}

// ---------------------------------------------------------------------------
extern "C" void kernel_launch(void* const* d_in, const int* in_sizes, int n_in,
                              void* d_out_, int out_size, void* d_ws,
                              size_t ws_size, hipStream_t stream) {
  (void)in_sizes; (void)n_in; (void)out_size; (void)ws_size;
  const float* X       = (const float*)d_in[0];
  const float* attn_nw = (const float*)d_in[1];
  const float* ffn_nw  = (const float*)d_in[2];
  const float* Wq      = (const float*)d_in[3];
  const float* Wdkv    = (const float*)d_in[4];
  const float* Wuk     = (const float*)d_in[5];
  const float* Wuv     = (const float*)d_in[6];
  const float* Wo      = (const float*)d_in[7];
  const float* Wr      = (const float*)d_in[8];
  const float* sh_w1   = (const float*)d_in[9];
  const float* sh_w3   = (const float*)d_in[10];
  const float* sh_w2   = (const float*)d_in[11];
  const float* rt_w1   = (const float*)d_in[12];
  const float* rt_w3   = (const float*)d_in[13];
  const float* rt_w2   = (const float*)d_in[14];

  float* out    = (float*)d_out_;
  float* outX   = out;                         // (1,2048,1024)
  float* outAux = out + (size_t)2048 * 1024;   // scalar
  float* outK   = outAux + 1;                  // (1,2048,16,64)
  float* outV   = outK + (size_t)2048 * 1024;  // (1,2048,16,64)

  char* wsb = (char*)d_ws;
  size_t off = 0;
  auto take_f = [&](size_t n) { float* p = (float*)(wsb + off); off += n * 4; return p; };
  auto take_h = [&](size_t n) {
    off = (off + 255) & ~(size_t)255;
    u16* p = (u16*)(wsb + off); off += n * 2; return p;
  };
  // f32 buffers
  float* X1  = take_f((size_t)2048 * 1024);
  float* xn2 = take_f((size_t)2048 * 1024);
  float* a1  = take_f((size_t)2048 * 512);
  float* a3  = take_f((size_t)2048 * 512);
  float* moe = take_f((size_t)2048 * 1024);
  float* gf  = take_f((size_t)2048 * 8);
  float* cnt = take_f(8);
  float* psm = take_f(8);     // contiguous with cnt
  // bf16 activation buffers
  u16* xn1b = take_h((size_t)2048 * 1024);
  u16* Qb   = take_h((size_t)2048 * 1024);
  u16* CKVb = take_h((size_t)2048 * 256);
  u16* Kbb  = take_h((size_t)2048 * 1024);
  u16* Vbb  = take_h((size_t)2048 * 1024);
  u16* attb = take_h((size_t)2048 * 1024);
  u16* xn2b = take_h((size_t)2048 * 1024);
  u16* hbb  = take_h((size_t)2048 * 512);
  // bf16 transposed weights [N][K]
  u16* WqT   = take_h((size_t)1024 * 1024);
  u16* WdkvT = take_h((size_t)1024 * 256);
  u16* WukT  = take_h((size_t)256 * 1024);
  u16* WuvT  = take_h((size_t)256 * 1024);
  u16* WoT   = take_h((size_t)1024 * 1024);
  u16* sh1T  = take_h((size_t)1024 * 512);
  u16* sh3T  = take_h((size_t)1024 * 512);
  u16* sh2T  = take_h((size_t)512 * 1024);
  u16* rt1T  = take_h((size_t)8 * 1024 * 512);
  u16* rt3T  = take_h((size_t)8 * 1024 * 512);
  u16* rt2T  = take_h((size_t)8 * 512 * 1024);

  hipMemsetAsync(cnt, 0, 16 * sizeof(float), stream);

  const dim3 b256(256);
  // ---- one-time (per launch) weight convert+transpose to bf16 ----
  transcvt_k<<<dim3(32, 32), b256, 0, stream>>>(Wq,   WqT,   1024, 1024);
  transcvt_k<<<dim3(8,  32), b256, 0, stream>>>(Wdkv, WdkvT, 1024, 256);
  transcvt_k<<<dim3(32, 8),  b256, 0, stream>>>(Wuk,  WukT,  256, 1024);
  transcvt_k<<<dim3(32, 8),  b256, 0, stream>>>(Wuv,  WuvT,  256, 1024);
  transcvt_k<<<dim3(32, 32), b256, 0, stream>>>(Wo,   WoT,   1024, 1024);
  transcvt_k<<<dim3(16, 32), b256, 0, stream>>>(sh_w1, sh1T, 1024, 512);
  transcvt_k<<<dim3(16, 32), b256, 0, stream>>>(sh_w3, sh3T, 1024, 512);
  transcvt_k<<<dim3(32, 16), b256, 0, stream>>>(sh_w2, sh2T, 512, 1024);
  transcvt_k<<<dim3(16, 32, 8), b256, 0, stream>>>(rt_w1, rt1T, 1024, 512);
  transcvt_k<<<dim3(16, 32, 8), b256, 0, stream>>>(rt_w3, rt3T, 1024, 512);
  transcvt_k<<<dim3(32, 16, 8), b256, 0, stream>>>(rt_w2, rt2T, 512, 1024);

  // ---- attention path ----
  rmsnorm_k<<<2048, 256, 0, stream>>>(X, attn_nw, nullptr, xn1b);
  gemm_k<0,1><<<dim3(16, 32), b256, 0, stream>>>(xn1b, WqT,   nullptr, Qb,   nullptr, nullptr, 0, 2048, 1024, 1024);
  gemm_k<0,1><<<dim3(4, 32),  b256, 0, stream>>>(xn1b, WdkvT, nullptr, CKVb, nullptr, nullptr, 0, 2048, 256, 1024);
  gemm_k<0,2><<<dim3(16, 32), b256, 0, stream>>>(CKVb, WukT,  outK, Kbb, nullptr, nullptr, 0, 2048, 1024, 256);
  gemm_k<0,2><<<dim3(16, 32), b256, 0, stream>>>(CKVb, WuvT,  outV, Vbb, nullptr, nullptr, 0, 2048, 1024, 256);
  flash_k<<<dim3(32, 16), dim3(128), 0, stream>>>(Qb, Kbb, Vbb, attb);
  gemm_k<1,0><<<dim3(16, 32), b256, 0, stream>>>(attb, WoT, X1, nullptr, X, nullptr, 0, 2048, 1024, 1024);
  // ---- MoE path ----
  rmsnorm_k<<<2048, 256, 0, stream>>>(X1, ffn_nw, xn2, xn2b);
  router_k<<<256, 256, 0, stream>>>(xn2, Wr, gf, cnt, psm);
  gemm_k<0,0><<<dim3(8, 32), b256, 0, stream>>>(xn2b, sh1T, a1, nullptr, nullptr, nullptr, 0, 2048, 512, 1024);
  gemm_k<0,0><<<dim3(8, 32), b256, 0, stream>>>(xn2b, sh3T, a3, nullptr, nullptr, nullptr, 0, 2048, 512, 1024);
  swiglu_k<<<4096, 256, 0, stream>>>(a1, a3, hbb, 2048 * 512);
  gemm_k<0,0><<<dim3(16, 32), b256, 0, stream>>>(hbb, sh2T, moe, nullptr, nullptr, nullptr, 0, 2048, 1024, 512);
  for (int e = 0; e < 8; ++e) {
    gemm_k<0,0><<<dim3(8, 32), b256, 0, stream>>>(xn2b, rt1T + (size_t)e * 1024 * 512, a1, nullptr, nullptr, nullptr, 0, 2048, 512, 1024);
    gemm_k<0,0><<<dim3(8, 32), b256, 0, stream>>>(xn2b, rt3T + (size_t)e * 1024 * 512, a3, nullptr, nullptr, nullptr, 0, 2048, 512, 1024);
    swiglu_k<<<4096, 256, 0, stream>>>(a1, a3, hbb, 2048 * 512);
    gemm_k<2,0><<<dim3(16, 32), b256, 0, stream>>>(hbb, rt2T + (size_t)e * 512 * 1024, moe, nullptr, nullptr, gf + e, 8, 2048, 1024, 512);
  }
  add_k<<<8192, 256, 0, stream>>>(X1, moe, outX, 2048 * 1024);
  aux_k<<<1, 32, 0, stream>>>(cnt, psm, outAux);
}
